// FusedMoEModularKernel_20899310863256
// MI455X (gfx1250) — compile-verified, matrix operand order checked by reading, not run on hardware
//
#include <hip/hip_runtime.h>
#include <hip/hip_bf16.h>

// ---------------------------------------------------------------------------
// Fused MoE for MI455X (gfx1250), wave32 + WMMA bf16 path.
//   M=1024 tokens, K=1024, E=8 experts, I=2048, N=2I, TOPK=2
// Bandwidth-bound: 196MB fp32 weights @23.3TB/s ~= 8.4us floor; sparse routing
// computes only the 2048 selected token-expert pairs (~26 GFLOP vs 103 dense).
// v2: double-buffered LDS + register staging (global loads for chunk i+1 in
// flight during WMMAs on chunk i), KC=64 -> 8 wmma per barrier per wave.
// ---------------------------------------------------------------------------

#define M_TOK 1024
#define K_DIM 1024
#define E_NUM 8
#define I_DIM 2048
#define TOPK  2
#define TILE  64
#define KC    64
#define LDS_STRIDE (KC + 8)                       // 72 bf16 = 144B rows (16B-aligned)
#define PAD_SLOTS  (M_TOK * TOPK + E_NUM * TILE)  // 2560
#define MAX_MT     ((M_TOK * TOPK) / TILE)        // 32 worst-case m-tiles/expert

typedef __bf16 v8bf  __attribute__((ext_vector_type(8)));
typedef __bf16 v16bf __attribute__((ext_vector_type(16)));
typedef float  v8f   __attribute__((ext_vector_type(8)));

// 16x32 bf16 WMMA operand fragment from LDS tile (row-major, LDS_STRIDE row
// pitch). Per ISA 16-bit A/B layout: lanes 0-15 hold K 0..7 & 16..23 of
// row=lane; lanes 16-31 hold K 8..15 & 24..31.  ks selects the 32-wide k-step.
__device__ inline v16bf load_frag(const __bf16* tileRowBase, int laneRow, int laneHalf, int ks) {
    const __bf16* row = tileRowBase + laneRow * LDS_STRIDE + ks * 32 + laneHalf * 8;
    v8bf lo = *reinterpret_cast<const v8bf*>(row);
    v8bf hi = *reinterpret_cast<const v8bf*>(row + 16);
    v16bf r;
#pragma unroll
    for (int i = 0; i < 8; ++i) { r[i] = lo[i]; r[i + 8] = hi[i]; }
    return r;
}

// Register staging: 16 contiguous fp32 (four float4) per thread per tile.
struct Stage { float4 f[4]; };

__device__ inline void ld_stage(Stage& s, const float* p) {
    const float4* q = reinterpret_cast<const float4*>(p);
    s.f[0] = q[0]; s.f[1] = q[1]; s.f[2] = q[2]; s.f[3] = q[3];
}

__device__ inline void st_stage(__bf16* dst, const Stage& s) {
    v8bf b0 = { (__bf16)s.f[0].x, (__bf16)s.f[0].y, (__bf16)s.f[0].z, (__bf16)s.f[0].w,
                (__bf16)s.f[1].x, (__bf16)s.f[1].y, (__bf16)s.f[1].z, (__bf16)s.f[1].w };
    v8bf b1 = { (__bf16)s.f[2].x, (__bf16)s.f[2].y, (__bf16)s.f[2].z, (__bf16)s.f[2].w,
                (__bf16)s.f[3].x, (__bf16)s.f[3].y, (__bf16)s.f[3].z, (__bf16)s.f[3].w };
    *reinterpret_cast<v8bf*>(dst)     = b0;
    *reinterpret_cast<v8bf*>(dst + 8) = b1;
}

// ---------------------------------------------------------------------------
__global__ void moe_zero(float* __restrict__ out, int n) {
    int i = blockIdx.x * blockDim.x + threadIdx.x;
    if (i < n) out[i] = 0.0f;
}

// Single-block routing: counts, tile-padded offsets, slot scatter.
__global__ void moe_route(const int* __restrict__ ids, const float* __restrict__ tw,
                          int* counts, int* fill, int* offs,
                          int* slot_token, float* slot_wt) {
    const int t = threadIdx.x;
    if (t < E_NUM) { counts[t] = 0; fill[t] = 0; }
    __threadfence();
    __syncthreads();
    for (int i = t; i < M_TOK * TOPK; i += blockDim.x)
        atomicAdd(&counts[ids[i]], 1);
    __threadfence();
    __syncthreads();
    if (t == 0) {
        int acc = 0;
        for (int e = 0; e < E_NUM; ++e) {
            offs[e] = acc;
            acc += (counts[e] + TILE - 1) / TILE * TILE;
        }
        offs[E_NUM] = acc;
    }
    __threadfence();
    __syncthreads();
    for (int i = t; i < M_TOK * TOPK; i += blockDim.x) {
        int e   = ids[i];
        int pos = atomicAdd(&fill[e], 1);
        int sl  = offs[e] + pos;
        slot_token[sl] = i / TOPK;
        slot_wt[sl]    = tw[i];
    }
}

// ---------------------------------------------------------------------------
// GEMM1 + SiLU*up. 64x64 gate & up tiles, 8 waves (2x4), 4 accums/wave,
// K chunks of 64 (2 wmma k-steps), double-buffered LDS.
__global__ __launch_bounds__(256)
void moe_gemm1_silu(const float* __restrict__ a1, const float* __restrict__ w1,
                    const int* __restrict__ counts, const int* __restrict__ offs,
                    const int* __restrict__ slot_token, __bf16* __restrict__ act) {
    const int e   = blockIdx.z;
    const int cnt = counts[e];
    const int mt  = blockIdx.y;
    if (mt * TILE >= cnt) return;
    const int base = offs[e];
    const int n0   = blockIdx.x * TILE;

    __shared__ int toks[TILE];
    __shared__ __align__(16) __bf16 As[2][TILE * LDS_STRIDE];
    __shared__ __align__(16) __bf16 Bg[2][TILE * LDS_STRIDE];
    __shared__ __align__(16) __bf16 Bu[2][TILE * LDS_STRIDE];

    const int t = threadIdx.x;
    if (t < TILE) {
        int local = mt * TILE + t;
        toks[t] = slot_token[base + (local < cnt ? local : 0)];
    }
    __syncthreads();

    const int lr  = t >> 2;              // tile row this thread stages (0..63)
    const int c16 = (t & 3) * 16;        // 16-wide column chunk
    const int wave = t >> 5, lane = t & 31;
    const int wm = wave >> 2, wn = wave & 3;        // 2x4 wave grid
    const int laneRow = lane & 15, laneHalf = lane >> 4;

    v8f accg0 = {}, accg1 = {}, accu0 = {}, accu1 = {};

    const float* aSrc = a1 + (size_t)toks[lr] * K_DIM + c16;
    const float* gSrc = w1 + ((size_t)e * (2 * I_DIM) + (n0 + lr)) * K_DIM + c16;
    const float* uSrc = w1 + ((size_t)e * (2 * I_DIM) + (I_DIM + n0 + lr)) * K_DIM + c16;

    Stage sa, sg, su;
    ld_stage(sa, aSrc); ld_stage(sg, gSrc); ld_stage(su, uSrc);
    int buf = 0;
    st_stage(&As[0][lr * LDS_STRIDE + c16], sa);
    st_stage(&Bg[0][lr * LDS_STRIDE + c16], sg);
    st_stage(&Bu[0][lr * LDS_STRIDE + c16], su);
    __syncthreads();

    for (int kb = 0; kb < K_DIM; kb += KC) {
        const int  nxt  = buf ^ 1;
        const bool more = (kb + KC) < K_DIM;
        if (more) {   // global loads for next chunk fly during this chunk's WMMAs
            ld_stage(sa, aSrc + kb + KC);
            ld_stage(sg, gSrc + kb + KC);
            ld_stage(su, uSrc + kb + KC);
        }
#pragma unroll
        for (int ks = 0; ks < 2; ++ks) {
            v16bf a0  = load_frag(&As[buf][(wm * 16)      * LDS_STRIDE], laneRow, laneHalf, ks);
            v16bf a1f = load_frag(&As[buf][(wm * 16 + 32) * LDS_STRIDE], laneRow, laneHalf, ks);
            v16bf bg  = load_frag(&Bg[buf][(wn * 16)      * LDS_STRIDE], laneRow, laneHalf, ks);
            v16bf bu  = load_frag(&Bu[buf][(wn * 16)      * LDS_STRIDE], laneRow, laneHalf, ks);
            accg0 = __builtin_amdgcn_wmma_f32_16x16x32_bf16(false, a0,  false, bg, (short)0, accg0, false, false);
            accg1 = __builtin_amdgcn_wmma_f32_16x16x32_bf16(false, a1f, false, bg, (short)0, accg1, false, false);
            accu0 = __builtin_amdgcn_wmma_f32_16x16x32_bf16(false, a0,  false, bu, (short)0, accu0, false, false);
            accu1 = __builtin_amdgcn_wmma_f32_16x16x32_bf16(false, a1f, false, bu, (short)0, accu1, false, false);
        }
        if (more) {
            st_stage(&As[nxt][lr * LDS_STRIDE + c16], sa);
            st_stage(&Bg[nxt][lr * LDS_STRIDE + c16], sg);
            st_stage(&Bu[nxt][lr * LDS_STRIDE + c16], su);
        }
        __syncthreads();
        buf = nxt;
    }

    // C/D layout: lanes 0-15 -> M=vgpr, N=lane; lanes 16-31 -> M=vgpr+8, N=lane-16
    const int mBase = mt * TILE;
#pragma unroll
    for (int s = 0; s < 2; ++s) {
        v8f g = s ? accg1 : accg0;
        v8f u = s ? accu1 : accu0;
        int mrow = wm * 16 + s * 32 + laneHalf * 8;
        int ncol = n0 + wn * 16 + laneRow;
#pragma unroll
        for (int v = 0; v < 8; ++v) {
            int ml = mrow + v;
            if (mBase + ml < cnt) {
                float gv = g[v], uv = u[v];
                float a  = gv / (1.0f + __expf(-gv)) * uv;  // silu(g)*u
                act[(size_t)(base + mBase + ml) * I_DIM + ncol] = (__bf16)a;
            }
        }
    }
}

// ---------------------------------------------------------------------------
// GEMM2 + weighted scatter: out[tok,k] += w_slot * (act[slot]·w2[e,k])
__global__ __launch_bounds__(256)
void moe_gemm2_scatter(const __bf16* __restrict__ act, const float* __restrict__ w2,
                       const int* __restrict__ counts, const int* __restrict__ offs,
                       const int* __restrict__ slot_token, const float* __restrict__ slot_wt,
                       float* __restrict__ out) {
    const int e   = blockIdx.z;
    const int cnt = counts[e];
    const int mt  = blockIdx.y;
    if (mt * TILE >= cnt) return;
    const int base = offs[e];
    const int k0   = blockIdx.x * TILE;

    __shared__ int   toks[TILE];
    __shared__ float wts[TILE];
    __shared__ __align__(16) __bf16 As[2][TILE * LDS_STRIDE];
    __shared__ __align__(16) __bf16 Bs[2][TILE * LDS_STRIDE];

    const int t = threadIdx.x;
    if (t < TILE) {
        int local = mt * TILE + t;
        int sl = base + (local < cnt ? local : 0);
        toks[t] = slot_token[sl];
        wts[t]  = slot_wt[sl];
    }
    __syncthreads();

    const int lr = t >> 2, c16 = (t & 3) * 16;
    const int wave = t >> 5, lane = t & 31;
    const int wm = wave >> 2, wn = wave & 3;
    const int laneRow = lane & 15, laneHalf = lane >> 4;

    v8f acc0 = {}, acc1 = {};
    const __bf16* aSrc = act + (size_t)(base + mt * TILE + lr) * I_DIM + c16;
    const float*  bSrc = w2 + ((size_t)e * K_DIM + (k0 + lr)) * I_DIM + c16;

    v8bf h0 = *reinterpret_cast<const v8bf*>(aSrc);
    v8bf h1 = *reinterpret_cast<const v8bf*>(aSrc + 8);
    Stage sb;
    ld_stage(sb, bSrc);
    int buf = 0;
    *reinterpret_cast<v8bf*>(&As[0][lr * LDS_STRIDE + c16])     = h0;
    *reinterpret_cast<v8bf*>(&As[0][lr * LDS_STRIDE + c16 + 8]) = h1;
    st_stage(&Bs[0][lr * LDS_STRIDE + c16], sb);
    __syncthreads();

    for (int ib = 0; ib < I_DIM; ib += KC) {
        const int  nxt  = buf ^ 1;
        const bool more = (ib + KC) < I_DIM;
        if (more) {
            h0 = *reinterpret_cast<const v8bf*>(aSrc + ib + KC);
            h1 = *reinterpret_cast<const v8bf*>(aSrc + ib + KC + 8);
            ld_stage(sb, bSrc + ib + KC);
        }
#pragma unroll
        for (int ks = 0; ks < 2; ++ks) {
            v16bf a0  = load_frag(&As[buf][(wm * 16)      * LDS_STRIDE], laneRow, laneHalf, ks);
            v16bf a1f = load_frag(&As[buf][(wm * 16 + 32) * LDS_STRIDE], laneRow, laneHalf, ks);
            v16bf b   = load_frag(&Bs[buf][(wn * 16)      * LDS_STRIDE], laneRow, laneHalf, ks);
            acc0 = __builtin_amdgcn_wmma_f32_16x16x32_bf16(false, a0,  false, b, (short)0, acc0, false, false);
            acc1 = __builtin_amdgcn_wmma_f32_16x16x32_bf16(false, a1f, false, b, (short)0, acc1, false, false);
        }
        if (more) {
            *reinterpret_cast<v8bf*>(&As[nxt][lr * LDS_STRIDE + c16])     = h0;
            *reinterpret_cast<v8bf*>(&As[nxt][lr * LDS_STRIDE + c16 + 8]) = h1;
            st_stage(&Bs[nxt][lr * LDS_STRIDE + c16], sb);
        }
        __syncthreads();
        buf = nxt;
    }

    const int mBase = mt * TILE;
#pragma unroll
    for (int s = 0; s < 2; ++s) {
        v8f acc = s ? acc1 : acc0;
        int mrow = wm * 16 + s * 32 + laneHalf * 8;
        int kcol = k0 + wn * 16 + laneRow;
#pragma unroll
        for (int v = 0; v < 8; ++v) {
            int ml = mrow + v;
            if (mBase + ml < cnt)
                atomicAdd(&out[(size_t)toks[ml] * K_DIM + kcol], acc[v] * wts[ml]);
        }
    }
}

// ---------------------------------------------------------------------------
extern "C" void kernel_launch(void* const* d_in, const int* in_sizes, int n_in,
                              void* d_out, int out_size, void* d_ws, size_t ws_size,
                              hipStream_t stream) {
    const float* a1  = (const float*)d_in[0];   // [M,K]
    const float* w1  = (const float*)d_in[1];   // [E,2I,K]
    const float* w2  = (const float*)d_in[2];   // [E,K,I]
    const float* tw  = (const float*)d_in[3];   // [M,TOPK]
    const int*   ids = (const int*)d_in[4];     // [M,TOPK]
    float* out = (float*)d_out;                 // [M,K]

    char* ws = (char*)d_ws;
    int*    counts     = (int*)(ws + 0);
    int*    fill       = (int*)(ws + 64);
    int*    offs       = (int*)(ws + 128);
    int*    slot_token = (int*)(ws + 256);
    float*  slot_wt    = (float*)(ws + 256 + PAD_SLOTS * sizeof(int));
    __bf16* act        = (__bf16*)(ws + 32768);   // PAD_SLOTS * I_DIM bf16 (~10.5MB)

    moe_zero<<<dim3((M_TOK * K_DIM + 1023) / 1024), dim3(1024), 0, stream>>>(out, M_TOK * K_DIM);
    moe_route<<<dim3(1), dim3(256), 0, stream>>>(ids, tw, counts, fill, offs, slot_token, slot_wt);
    moe_gemm1_silu<<<dim3(I_DIM / TILE, MAX_MT, E_NUM), dim3(256), 0, stream>>>(
        a1, w1, counts, offs, slot_token, act);
    moe_gemm2_scatter<<<dim3(K_DIM / TILE, MAX_MT, E_NUM), dim3(256), 0, stream>>>(
        act, w2, counts, offs, slot_token, slot_wt, out);
}
